// SpatialAttention_18013092840142
// MI455X (gfx1250) — matrix operand order repaired and checked by reading, not compiled, not run on hardware
//
#include <hip/hip_runtime.h>
#include <hip/hip_bf16.h>

// ---------------------------------------------------------------------------
// SpatialAttention (Linformer-style) on gfx1250 via v_wmma_f32_16x16x32_bf16.
// All contractions go through WMMA (f32 accumulate); intermediates are bf16 in
// workspace (~128 MB). Softmax fused with score/PV GEMMs per 16-row tile.
// Every WMMA operand is a contiguous 16/32-byte load (k/v pre-transposed).
// q rows are staged to LDS with gfx1250 async loads (global_load_async_to_lds).
// ---------------------------------------------------------------------------

#define BN    4
#define SEQ   4096
#define CDIM  512
#define NHEAD 8
#define HDIM  64
#define PDIM  512
#define QKVC  1536   // 3*CDIM
#define HEADS (BN*NHEAD)

typedef __attribute__((ext_vector_type(16))) __bf16 v16bf;
typedef __attribute__((ext_vector_type(8)))  float  v8f;

union Frag {
    v16bf v;
    unsigned short s[16];
    uint4 q[2];
};

__device__ __forceinline__ unsigned short f2bf(float f) {
    union { float f; unsigned int u; } c; c.f = f;
    unsigned int u = c.u;
    u += 0x7FFFu + ((u >> 16) & 1u);   // round-to-nearest-even
    return (unsigned short)(u >> 16);
}
__device__ __forceinline__ float bf2f(unsigned short h) {
    union { unsigned int u; float f; } c; c.u = ((unsigned int)h) << 16;
    return c.f;
}

// A-fragment (16x32 bf16, MxK): lane group g holds K runs {g*8..+7} and {16+g*8..+7}
__device__ __forceinline__ void load_a_frag(Frag& f, const unsigned short* row, int k0, int g) {
    f.q[0] = *(const uint4*)(row + k0 + g * 8);
    f.q[1] = *(const uint4*)(row + k0 + 16 + g * 8);
}
// B-fragment (32x16 bf16, KxN): lane = column, group g holds K runs {g*8..+7} and {16+g*8..+7}
__device__ __forceinline__ void load_b_frag(Frag& f, const unsigned short* col, int k0, int g) {
    f.q[0] = *(const uint4*)(col + k0 + g * 8);
    f.q[1] = *(const uint4*)(col + k0 + 16 + g * 8);
}
__device__ __forceinline__ v8f wmma_bf16(const Frag& a, const Frag& b, v8f c) {
    return __builtin_amdgcn_wmma_f32_16x16x32_bf16(false, a.v, false, b.v,
                                                   (short)0, c, false, false);
}

// ---------------------------------------------------------------------------
// K0: f32 -> bf16 conversion
// ---------------------------------------------------------------------------
__global__ void cvt_f32_bf16(const float* __restrict__ in,
                             unsigned short* __restrict__ out, int n) {
    int i = blockIdx.x * 256 + threadIdx.x;
    if (i < n) out[i] = f2bf(in[i]);
}

// ---------------------------------------------------------------------------
// K1 / K6: generic WMMA GEMM  C[m][n] = sum_k A[m][k] * Bw[n][k] (+ bias[n])
// block = 128 (4 waves); wave -> 16-col subtile. grid = (M/16, N/64).
// ---------------------------------------------------------------------------
__global__ void gemm_bf16(const unsigned short* __restrict__ A,
                          const unsigned short* __restrict__ Bw,
                          const float* __restrict__ bias,
                          unsigned short* __restrict__ Cb,   // bf16 out (or null)
                          float* __restrict__ Cf,            // f32 out (or null)
                          int M, int N, int K) {
    int wave = threadIdx.x >> 5;
    int lane = threadIdx.x & 31;
    int l15 = lane & 15, g = lane >> 4;
    int mbase = blockIdx.x * 16;
    int nbase = blockIdx.y * 64 + wave * 16;

    const unsigned short* arow = A + (size_t)(mbase + l15) * K;
    const unsigned short* brow = Bw + (size_t)(nbase + l15) * K;

    v8f acc = {};
    for (int k0 = 0; k0 < K; k0 += 32) {
        Frag a, b;
        load_a_frag(a, arow, k0, g);
        load_b_frag(b, brow, k0, g);
        __builtin_prefetch(arow + k0 + 256, 0, 1);   // global_prefetch_b8
        __builtin_prefetch(brow + k0 + 256, 0, 1);
        acc = wmma_bf16(a, b, acc);
    }
    int col = nbase + l15;
    float bv = bias ? bias[col] : 0.0f;
    for (int v = 0; v < 8; ++v) {
        int row = mbase + v + 8 * g;
        float val = acc[v] + bv;
        if (Cb) Cb[(size_t)row * N + col] = f2bf(val);
        else    Cf[(size_t)row * N + col] = val;
    }
}

// ---------------------------------------------------------------------------
// K2: per-(b,h,d) L2 norm of q over the sequence axis -> 1/max(||.||, eps)
// one wave per (b,h,d). block=256 (8 waves), grid=256.
// ---------------------------------------------------------------------------
__global__ void qscale_kernel(const unsigned short* __restrict__ qkv,
                              float* __restrict__ qscale) {
    int wave = threadIdx.x >> 5;
    int lane = threadIdx.x & 31;
    int flat = blockIdx.x * 8 + wave;          // b*512 + h*64 + d
    int b = flat >> 9;
    int col = flat & 511;                      // q columns start at 0
    const unsigned short* base = qkv + ((size_t)b * SEQ) * QKVC + col;
    float s = 0.f;
    for (int i = 0; i < SEQ / 32; ++i) {
        float f = bf2f(base[(size_t)(lane + i * 32) * QKVC]);
        s += f * f;
    }
    for (int m = 16; m >= 1; m >>= 1) s += __shfl_xor(s, m, 32);
    if (lane == 0) qscale[flat] = 1.0f / fmaxf(sqrtf(s), 1e-12f);
}

// ---------------------------------------------------------------------------
// K3: transpose k,v into per-head [d][n] layout (coalesced writes) so the
// E-projection GEMM sees K-contiguous A rows.
// grid = (SEQ/256, HDIM, HEADS), block = 256.
// ---------------------------------------------------------------------------
__global__ void kv_transpose(const unsigned short* __restrict__ qkv,
                             unsigned short* __restrict__ kT,
                             unsigned short* __restrict__ vT) {
    int n = blockIdx.x * 256 + threadIdx.x;
    int d = blockIdx.y;
    int head = blockIdx.z;
    int b = head >> 3, h = head & 7;
    size_t src = ((size_t)(b * SEQ + n)) * QKVC + h * HDIM + d;
    size_t dst = ((size_t)head * HDIM + d) * SEQ + n;
    kT[dst] = qkv[src + 512];
    vT[dst] = qkv[src + 1024];
}

// ---------------------------------------------------------------------------
// K4: per-head projections over K = SEQ = 4096 (all-contiguous operands):
//   kpT[head][p][d] = (sum_n kT[d][n]*E_w[p][n] + E_b[p]) * qscale[d]
//   vpP[head][d][p] =  sum_n vT[d][n]*E_w[p][n] + E_b[p]
// A = kT/vT rows (d), B = E_w rows (p), C[m=d][n=p].
// block = 128 (4 waves over p); grid = (HDIM/16, PDIM/64, HEADS).
// ---------------------------------------------------------------------------
__global__ void proj_kernel(const unsigned short* __restrict__ kT,
                            const unsigned short* __restrict__ vT,
                            const unsigned short* __restrict__ Ew,
                            const float* __restrict__ Eb,
                            const float* __restrict__ qscale,
                            unsigned short* __restrict__ kpT,
                            unsigned short* __restrict__ vpP) {
    int wave = threadIdx.x >> 5;
    int lane = threadIdx.x & 31;
    int l15 = lane & 15, g = lane >> 4;
    int dbase = blockIdx.x * 16;
    int pbase = blockIdx.y * 64 + wave * 16;
    int head = blockIdx.z;

    const unsigned short* akrow = kT + ((size_t)head * HDIM + dbase + l15) * SEQ;
    const unsigned short* avrow = vT + ((size_t)head * HDIM + dbase + l15) * SEQ;
    const unsigned short* brow  = Ew + (size_t)(pbase + l15) * SEQ;

    v8f acck = {}, accv = {};
    for (int k0 = 0; k0 < SEQ; k0 += 32) {
        Frag ak, av, be;
        load_a_frag(ak, akrow, k0, g);
        load_a_frag(av, avrow, k0, g);
        load_b_frag(be, brow, k0, g);
        __builtin_prefetch(akrow + k0 + 256, 0, 1);
        __builtin_prefetch(avrow + k0 + 256, 0, 1);
        __builtin_prefetch(brow  + k0 + 256, 0, 1);
        acck = wmma_bf16(ak, be, acck);
        accv = wmma_bf16(av, be, accv);
    }
    int p = pbase + l15;
    float eb = Eb[p];
    for (int v = 0; v < 8; ++v) {
        int d = dbase + v + 8 * g;
        float sq = qscale[head * HDIM + d];
        kpT[((size_t)head * PDIM + p) * HDIM + d] = f2bf((acck[v] + eb) * sq);
        vpP[((size_t)head * HDIM + d) * PDIM + p] = f2bf(accv[v] + eb);
    }
}

// ---------------------------------------------------------------------------
// K5: fused scores + softmax + PV.
// block = 256 (8 waves) per (head, 16-row n-tile).
//  - q rows (16x64 bf16) async-staged to LDS (global_load_async_to_lds_b128)
//  - Phase 1: each wave -> 4 p-subtiles of scores (full 16x512 in accumulators)
//  - Phase 2: cross-wave max/sum softmax; normalized bf16 scores in LDS
//  - Phase 3: WMMA vs vpP (contiguous B), write torch-faithful permuted layout
// ---------------------------------------------------------------------------
__global__ void attn_kernel(const unsigned short* __restrict__ qkv,
                            const unsigned short* __restrict__ kpT,
                            const unsigned short* __restrict__ vpP,
                            unsigned short* __restrict__ xsab) {
    __shared__ float red[8][16];
    __shared__ float rowmax[16], rowsum[16];
    __shared__ unsigned short qtile[16][HDIM];   // 2 KB
    __shared__ unsigned short attnb[16][PDIM];   // 16 KB

    int wave = threadIdx.x >> 5;
    int lane = threadIdx.x & 31;
    int l15 = lane & 15, g = lane >> 4;
    int nbase = blockIdx.x * 16;
    int head = blockIdx.y;
    int b = head >> 3, h = head & 7;

    const unsigned short* kph = kpT + (size_t)head * PDIM * HDIM;
    const unsigned short* vph = vpP + (size_t)head * HDIM * PDIM;

    // ---- async-stage q rows to LDS (CDNA5 async load path) ---------------
    {
        int t = threadIdx.x;
        if (t < 128) {
            int r = t >> 3, cch = (t & 7) * 8;                 // 8 bf16 = 16 B
            const unsigned short* gp =
                qkv + ((size_t)(b * SEQ + nbase + r)) * QKVC + h * HDIM + cch;
            unsigned ldsoff = (unsigned)(unsigned long long)(&qtile[r][cch]);
            asm volatile("global_load_async_to_lds_b128 %0, %1, off"
                         :: "v"(ldsoff), "v"(gp) : "memory");
        }
        asm volatile("s_wait_asynccnt 0x0" ::: "memory");
        __syncthreads();
    }

    // ---- Phase 1: scores -------------------------------------------------
    v8f acc[4];
    for (int a = 0; a < 4; ++a) {
        int pt = wave * 4 + a;
        const unsigned short* brow = kph + (size_t)(pt * 16 + l15) * HDIM;
        v8f c = {};
        for (int k0 = 0; k0 < HDIM; k0 += 32) {
            Frag af, bf;
            load_a_frag(af, &qtile[l15][0], k0, g);
            load_b_frag(bf, brow, k0, g);
            c = wmma_bf16(af, bf, c);
        }
        acc[a] = c;
    }

    // ---- Phase 2: softmax ------------------------------------------------
    for (int v = 0; v < 8; ++v) {
        float m = fmaxf(fmaxf(acc[0][v], acc[1][v]), fmaxf(acc[2][v], acc[3][v]));
        for (int s = 8; s >= 1; s >>= 1) m = fmaxf(m, __shfl_xor(m, s, 32));
        if (l15 == 0) red[wave][v + 8 * g] = m;
    }
    __syncthreads();
    if (threadIdx.x < 16) {
        float m = red[0][threadIdx.x];
        for (int w = 1; w < 8; ++w) m = fmaxf(m, red[w][threadIdx.x]);
        rowmax[threadIdx.x] = m;
    }
    __syncthreads();
    for (int v = 0; v < 8; ++v) {
        float rm = rowmax[v + 8 * g];
        float s = 0.f;
        for (int a = 0; a < 4; ++a) {
            float e = __expf(acc[a][v] - rm);
            acc[a][v] = e;
            s += e;
        }
        for (int sh = 8; sh >= 1; sh >>= 1) s += __shfl_xor(s, sh, 32);
        if (l15 == 0) red[wave][v + 8 * g] = s;
    }
    __syncthreads();
    if (threadIdx.x < 16) {
        float s = red[0][threadIdx.x];
        for (int w = 1; w < 8; ++w) s += red[w][threadIdx.x];
        rowsum[threadIdx.x] = s;
    }
    __syncthreads();
    for (int v = 0; v < 8; ++v) {
        int row = v + 8 * g;
        float inv = 1.0f / rowsum[row];
        for (int a = 0; a < 4; ++a)
            attnb[row][wave * 64 + a * 16 + l15] = f2bf(acc[a][v] * inv);
    }
    __syncthreads();

    // ---- Phase 3: PV (B contiguous from vpP[d][p]) -----------------------
    int dt = wave & 3;                 // waves 4..7 duplicate; EXEC stays full
    int dcol = dt * 16 + l15;
    const unsigned short* brow = vph + (size_t)dcol * PDIM;
    v8f c = {};
    for (int k0 = 0; k0 < PDIM; k0 += 32) {
        Frag af, bf;
        load_a_frag(af, &attnb[l15][0], k0, g);
        load_b_frag(bf, brow, k0, g);
        c = wmma_bf16(af, bf, c);
    }
    if (wave < 4) {
        // torch-faithful permute: x_sa[b,h,n,d] -> row n' = d*64 + h*8 + n/512,
        // col c' = n % 512   (flat [B,HD,NH,N] reshaped to [B,N,C])
        for (int v = 0; v < 8; ++v) {
            int n = nbase + v + 8 * g;
            int nprime = dcol * 64 + h * 8 + (n >> 9);
            int cprime = n & 511;
            xsab[(((size_t)b * SEQ) + nprime) * CDIM + cprime] = f2bf(c[v]);
        }
    }
}

// ---------------------------------------------------------------------------
// Host launcher
// ---------------------------------------------------------------------------
extern "C" void kernel_launch(void* const* d_in, const int* in_sizes, int n_in,
                              void* d_out, int out_size, void* d_ws, size_t ws_size,
                              hipStream_t stream) {
    const float* x      = (const float*)d_in[0];   // [4,4096,512]
    const float* qkv_w  = (const float*)d_in[1];   // [1536,512]
    const float* E_w    = (const float*)d_in[2];   // [512,4096]
    const float* E_b    = (const float*)d_in[3];   // [512]
    const float* out_w  = (const float*)d_in[4];   // [512,512]
    const float* out_b  = (const float*)d_in[5];   // [512]
    float* out = (float*)d_out;                    // [4,4096,512]

    // workspace carve-up (bytes); ~128 MB total, 256-B aligned by construction
    char* ws = (char*)d_ws;
    unsigned short* x_bf    = (unsigned short*)(ws);               // 16 MB
    unsigned short* wqkv_bf = (unsigned short*)(ws + 16777216);    // 1.5 MB
    unsigned short* Ew_bf   = (unsigned short*)(ws + 18350080);    // 4 MB
    unsigned short* outw_bf = (unsigned short*)(ws + 22544384);    // 0.5 MB
    unsigned short* qkv_bf  = (unsigned short*)(ws + 23068672);    // 48 MB
    float*          qscale  = (float*)         (ws + 73400320);    // 8 KB
    unsigned short* kT      = (unsigned short*)(ws + 73408512);    // 16 MB
    unsigned short* vT      = (unsigned short*)(ws + 90185728);    // 16 MB
    unsigned short* kpT     = (unsigned short*)(ws + 106962944);   // 2 MB
    unsigned short* vpP     = (unsigned short*)(ws + 109060096);   // 2 MB
    unsigned short* xsab    = (unsigned short*)(ws + 111157248);   // 16 MB

    // K0: conversions
    cvt_f32_bf16<<<(BN*SEQ*CDIM + 255) / 256, 256, 0, stream>>>(x, x_bf, BN*SEQ*CDIM);
    cvt_f32_bf16<<<(QKVC*CDIM   + 255) / 256, 256, 0, stream>>>(qkv_w, wqkv_bf, QKVC*CDIM);
    cvt_f32_bf16<<<(PDIM*SEQ    + 255) / 256, 256, 0, stream>>>(E_w, Ew_bf, PDIM*SEQ);
    cvt_f32_bf16<<<(CDIM*CDIM   + 255) / 256, 256, 0, stream>>>(out_w, outw_bf, CDIM*CDIM);

    // K1: qkv = x @ qkv_w^T -> bf16 [16384,1536]
    gemm_bf16<<<dim3(BN*SEQ/16, QKVC/64), 128, 0, stream>>>(
        x_bf, wqkv_bf, nullptr, qkv_bf, nullptr, BN*SEQ, QKVC, CDIM);

    // K2: q sequence-axis L2 norms -> scales
    qscale_kernel<<<BN*NHEAD*HDIM/8, 256, 0, stream>>>(qkv_bf, qscale);

    // K3: k/v transpose to per-head [d][n]
    kv_transpose<<<dim3(SEQ/256, HDIM, HEADS), 256, 0, stream>>>(qkv_bf, kT, vT);

    // K4: per-head k/v sequence projections (+bias; k folded with q-norm scale)
    proj_kernel<<<dim3(HDIM/16, PDIM/64, HEADS), 128, 0, stream>>>(
        kT, vT, Ew_bf, E_b, qscale, kpT, vpP);

    // K5: fused scores + softmax + PV, permuted bf16 output
    attn_kernel<<<dim3(SEQ/16, HEADS), 256, 0, stream>>>(qkv_bf, kpT, vpP, xsab);

    // K6: final projection + bias -> f32 d_out
    gemm_bf16<<<dim3(BN*SEQ/16, CDIM/64), 128, 0, stream>>>(
        xsab, outw_bf, out_b, nullptr, out, BN*SEQ, CDIM, CDIM);
}
